// ScaledDotProductAttention_8117488189582
// MI455X (gfx1250) — matrix operand order repaired and checked by reading, not compiled, not run on hardware
//
#include <hip/hip_runtime.h>

// ---------------------------------------------------------------------------
// Types
// ---------------------------------------------------------------------------
typedef unsigned short ushort_t;
typedef __attribute__((ext_vector_type(8)))  unsigned short u16x8;
typedef __attribute__((ext_vector_type(16))) __bf16         v16bf;
typedef __attribute__((ext_vector_type(8)))  float          v8f;

union FragBF16 {
    v16bf v;
    u16x8 h[2];
};

// Hardware f32->bf16 (RNE) via native conversion.
__device__ __forceinline__ ushort_t f32_to_bf16(float x) {
    __bf16 b = (__bf16)x;
    return __builtin_bit_cast(unsigned short, b);
}
__device__ __forceinline__ float bf16_to_f32(ushort_t h) {
    __bf16 b = __builtin_bit_cast(__bf16, h);
    return (float)b;
}

// ---------------------------------------------------------------------------
// LDS tile geometry: 128 rows x 32 bf16, rows padded to 40 elems (80 B) so
// consecutive rows land 20 banks apart (16-row period in the 64 banks).
// ---------------------------------------------------------------------------
#define TILE_ROWS   128
#define TILE_K      32
#define ROW_PITCH   40                        // elements (80 bytes)
#define TILE_ELEMS  (TILE_ROWS * ROW_PITCH)   // 5120 elements = 10 KB
#define TILE_BYTES  (TILE_ELEMS * 2)          // 10240

// Issue the 8 per-thread async 16B copies for one k-step (A tile + B tile).
// Pointers are pre-computed per thread and bumped by the caller.
__device__ __forceinline__ void issue_stage(const ushort_t* const* pA,
                                            const unsigned* laA, unsigned baseA,
                                            const ushort_t* const* pB,
                                            const unsigned* laB, unsigned baseB) {
#pragma unroll
    for (int i = 0; i < 4; ++i) {
        asm volatile("global_load_async_to_lds_b128 %0, %1, off"
                     :: "v"(baseA + laA[i]),
                        "v"((unsigned long long)(size_t)pA[i])
                     : "memory");
    }
#pragma unroll
    for (int i = 0; i < 4; ++i) {
        asm volatile("global_load_async_to_lds_b128 %0, %1, off"
                     :: "v"(baseB + laB[i]),
                        "v"((unsigned long long)(size_t)pB[i])
                     : "memory");
    }
}

// ---------------------------------------------------------------------------
// GEMM:  D[b] = alpha * A[b] (MxK) * W[b] (NxK)^T
// Block: 128 threads = 4 waves (2x2); wave tile 64x64 (4x4 WMMA tiles);
// block tile 128x128. K consumed in steps of 32 through double-buffered LDS
// (async global->LDS copies, ASYNCcnt synchronized).
//   OUT_F32 = 0 -> bf16 output (tstore==1 stores transposed: Out[n][m])
//   OUT_F32 = 1 -> f32 output
// All shapes used here divide evenly -> no bounds checks.
// ---------------------------------------------------------------------------
template<int OUT_F32>
__global__ __launch_bounds__(128)
void gemm_nt_bf16(const ushort_t* __restrict__ A, int lda, long long strideA,
                  const ushort_t* __restrict__ W, int ldw, long long strideW,
                  void* __restrict__ Out, int ldo, long long strideO,
                  int K, float alpha, int tstore) {
    __shared__ ushort_t smem[2][2][TILE_ELEMS];   // [buf][A=0/B=1][...], 40 KB

    const int t    = threadIdx.x;
    const int lane = t & 31;
    const int wave = t >> 5;
    const int wm   = wave >> 1;              // 0..1
    const int wn   = wave & 1;               // 0..1
    const int bmb  = blockIdx.y * 128;       // block row origin
    const int bnb  = blockIdx.x * 128;       // block col origin

    const ushort_t* Ab = A + (size_t)blockIdx.z * strideA;
    const ushort_t* Wb = W + (size_t)blockIdx.z * strideW;

    // LDS byte offsets of the four tiles (smem assumed at LDS offset 0).
    const unsigned ldsA0 = 0;
    const unsigned ldsB0 = TILE_BYTES;
    const unsigned ldsA1 = 2 * TILE_BYTES;
    const unsigned ldsB1 = 3 * TILE_BYTES;

    // Per-thread staging addresses: chunk c = t + 128*i ->
    //   row = (t>>2) + 32*i (c16 = t&3 is i-invariant). Bump by TILE_K elems/step.
    const int rb  = t >> 2;
    const int c16 = t & 3;
    const ushort_t* pA[4];
    const ushort_t* pB[4];
    unsigned laA[4], laB[4];
#pragma unroll
    for (int i = 0; i < 4; ++i) {
        const int row = rb + 32 * i;
        pA[i]  = Ab + (size_t)(bmb + row) * lda + c16 * 8;
        pB[i]  = Wb + (size_t)(bnb + row) * ldw + c16 * 8;
        laA[i] = (unsigned)(row * (ROW_PITCH * 2) + c16 * 16);
        laB[i] = laA[i];
    }

    // Prologue: stage k-step 0 into buffer 0.
    issue_stage(pA, laA, ldsA0, pB, laB, ldsB0);
#pragma unroll
    for (int i = 0; i < 4; ++i) { pA[i] += TILE_K; pB[i] += TILE_K; }

    v8f acc[4][4] = {};

    const int KT = K >> 5;
    for (int kt = 0; kt < KT; ++kt) {
        asm volatile("s_wait_asynccnt 0x0" ::: "memory");
        __syncthreads();

        const int cur = kt & 1;
        if (kt + 1 < KT) {
            issue_stage(pA, laA, cur ? ldsA0 : ldsA1,
                        pB, laB, cur ? ldsB0 : ldsB1);
#pragma unroll
            for (int i = 0; i < 4; ++i) { pA[i] += TILE_K; pB[i] += TILE_K; }
        }

        const ushort_t* As = &smem[cur][0][0];
        const ushort_t* Bs = &smem[cur][1][0];

        // Fragment reads (contiguous k = 16*(lane>>4) + s in both A and B).
        v16bf af[4], bf[4];
        const int kh  = (lane >> 4) << 4;    // A: 0 or 16
        const int k2  = (lane >> 4) << 4;    // B: 0 or 16
        const int l15 = lane & 15;
#pragma unroll
        for (int i = 0; i < 4; ++i) {
            const ushort_t* pa = As + (wm * 64 + i * 16 + l15) * ROW_PITCH + kh;
            FragBF16 fa;
            fa.h[0] = *(const u16x8*)(pa);
            fa.h[1] = *(const u16x8*)(pa + 8);
            af[i] = fa.v;
            const ushort_t* pb = Bs + (wn * 64 + i * 16 + l15) * ROW_PITCH + k2;
            FragBF16 fb;
            fb.h[0] = *(const u16x8*)(pb);
            fb.h[1] = *(const u16x8*)(pb + 8);
            bf[i] = fb.v;
        }

#pragma unroll
        for (int i = 0; i < 4; ++i)
#pragma unroll
            for (int j = 0; j < 4; ++j)
                acc[i][j] = __builtin_amdgcn_wmma_f32_16x16x32_bf16(
                    false, af[i], false, bf[j], (short)0, acc[i][j], false, false);
    }

    // Epilogue. C/D layout: lane (L&15)=column, VGPR r = row (L>>4)*8 + r.
    const int ncol  = lane & 15;
    const int mrow0 = (lane >> 4) * 8;
#pragma unroll
    for (int i = 0; i < 4; ++i) {
#pragma unroll
        for (int j = 0; j < 4; ++j) {
            const int mbase = bmb + wm * 64 + i * 16 + mrow0;
            const int n     = bnb + wn * 64 + j * 16 + ncol;
#pragma unroll
            for (int r = 0; r < 8; ++r) {
                const int m = mbase + r;
                const float val = alpha * acc[i][j][r];
                if (OUT_F32) {
                    float* O = (float*)Out + (size_t)blockIdx.z * strideO;
                    O[(size_t)m * ldo + n] = val;
                } else if (tstore) {
                    ushort_t* O = (ushort_t*)Out + (size_t)blockIdx.z * strideO;
                    O[(size_t)n * ldo + m] = f32_to_bf16(val);
                } else {
                    ushort_t* O = (ushort_t*)Out + (size_t)blockIdx.z * strideO;
                    O[(size_t)m * ldo + n] = f32_to_bf16(val);
                }
            }
        }
    }
}

// ---------------------------------------------------------------------------
// f32 -> bf16 bulk convert (8 elems / thread)
// ---------------------------------------------------------------------------
__global__ __launch_bounds__(256)
void cvt_f32_to_bf16(const float* __restrict__ src, ushort_t* __restrict__ dst,
                     long long n8) {
    long long i = (long long)blockIdx.x * blockDim.x + threadIdx.x;
    if (i >= n8) return;
    const float4* s = (const float4*)(src + i * 8);
    float4 lo = s[0];
    float4 hi = s[1];
    u16x8 o;
    o[0] = f32_to_bf16(lo.x); o[1] = f32_to_bf16(lo.y);
    o[2] = f32_to_bf16(lo.z); o[3] = f32_to_bf16(lo.w);
    o[4] = f32_to_bf16(hi.x); o[5] = f32_to_bf16(hi.y);
    o[6] = f32_to_bf16(hi.z); o[7] = f32_to_bf16(hi.w);
    *(u16x8*)(dst + i * 8) = o;
}

// ---------------------------------------------------------------------------
// In-place softmax over rows (ncols == 2048) of a bf16 matrix.
// One 256-thread block per row; values live in registers between passes.
// ---------------------------------------------------------------------------
__global__ __launch_bounds__(256)
void softmax_rows_bf16(ushort_t* __restrict__ S) {
    const int row = blockIdx.x;
    const int t   = threadIdx.x;
    ushort_t* p = S + (size_t)row * 2048 + t * 8;

    u16x8 raw = *(const u16x8*)p;
    float v[8];
    float mx = -3.402823466e38f;
#pragma unroll
    for (int i = 0; i < 8; ++i) {
        v[i] = bf16_to_f32(raw[i]);
        mx = fmaxf(mx, v[i]);
    }

    __shared__ float red[256];
    red[t] = mx;
    __syncthreads();
    for (int s = 128; s > 0; s >>= 1) {
        if (t < s) red[t] = fmaxf(red[t], red[t + s]);
        __syncthreads();
    }
    mx = red[0];
    __syncthreads();

    float sum = 0.0f;
#pragma unroll
    for (int i = 0; i < 8; ++i) {
        v[i] = __expf(v[i] - mx);
        sum += v[i];
    }
    red[t] = sum;
    __syncthreads();
    for (int s = 128; s > 0; s >>= 1) {
        if (t < s) red[t] += red[t + s];
        __syncthreads();
    }
    const float inv = 1.0f / red[0];

    u16x8 o;
#pragma unroll
    for (int i = 0; i < 8; ++i) o[i] = f32_to_bf16(v[i] * inv);
    *(u16x8*)p = o;
}

// ---------------------------------------------------------------------------
// Host-side orchestration
// ---------------------------------------------------------------------------
extern "C" void kernel_launch(void* const* d_in, const int* in_sizes, int n_in,
                              void* d_out, int out_size, void* d_ws, size_t ws_size,
                              hipStream_t stream) {
    (void)in_sizes; (void)n_in; (void)out_size; (void)ws_size;

    const float* x  = (const float*)d_in[0];  // [4,2048,1024]
    const float* wq = (const float*)d_in[1];  // [1024,1024] (out,in)
    const float* wk = (const float*)d_in[2];
    const float* wv = (const float*)d_in[3];
    const float* wo = (const float*)d_in[4];
    float* out = (float*)d_out;               // [4,2048,1024]

    const size_t P   = 8192ull * 1024;        // x / Q / K / V / C element count
    const size_t Wsz = 1024ull * 1024;        // weight element count
    const size_t Ssz = 4ull * 2048 * 2048;    // scores/probs element count

    // Workspace layout (bf16 = ushort): 5P + 4Wsz + Ssz elems ~= 120 MB.
    ushort_t* xb  = (ushort_t*)d_ws;
    ushort_t* wqb = xb  + P;
    ushort_t* wkb = wqb + Wsz;
    ushort_t* wvb = wkb + Wsz;
    ushort_t* wob = wvb + Wsz;
    ushort_t* Qb  = wob + Wsz;
    ushort_t* Kb  = Qb  + P;
    ushort_t* Vtb = Kb  + P;                  // V stored transposed: [b][1024][2048]
    ushort_t* Sb  = Vtb + P;                  // scores -> probs, bf16, [b][2048][2048]
    ushort_t* Cb  = Sb  + Ssz;                // attention context bf16 [b][2048][1024]

    const long long SEQ_D  = 2048ll * 1024;   // per-batch x/Q/K/C stride
    const long long D_SEQ  = 1024ll * 2048;   // per-batch Vt stride
    const long long SEQ_SQ = 2048ll * 2048;   // per-batch scores stride

    // --- 1. fp32 -> bf16 converts ---
    cvt_f32_to_bf16<<<(unsigned)((P / 8 + 255) / 256), 256, 0, stream>>>(x,  xb,  (long long)(P / 8));
    cvt_f32_to_bf16<<<(unsigned)((Wsz / 8 + 255) / 256), 256, 0, stream>>>(wq, wqb, (long long)(Wsz / 8));
    cvt_f32_to_bf16<<<(unsigned)((Wsz / 8 + 255) / 256), 256, 0, stream>>>(wk, wkb, (long long)(Wsz / 8));
    cvt_f32_to_bf16<<<(unsigned)((Wsz / 8 + 255) / 256), 256, 0, stream>>>(wv, wvb, (long long)(Wsz / 8));
    cvt_f32_to_bf16<<<(unsigned)((Wsz / 8 + 255) / 256), 256, 0, stream>>>(wo, wob, (long long)(Wsz / 8));

    const dim3 blk(128);
    // grid = (N/128, M/128, batch)
    const dim3 gProj(1024 / 128, 2048 / 128, 4);   // M=2048, N=1024 per batch
    const dim3 gScore(2048 / 128, 2048 / 128, 4);  // M=2048, N=2048 per batch

    // --- 2. Q / K projections: Q = x @ Wq^T ---
    gemm_nt_bf16<0><<<gProj, blk, 0, stream>>>(xb, 1024, SEQ_D, wqb, 1024, 0,
                                               Qb, 1024, SEQ_D, 1024, 1.0f, 0);
    gemm_nt_bf16<0><<<gProj, blk, 0, stream>>>(xb, 1024, SEQ_D, wkb, 1024, 0,
                                               Kb, 1024, SEQ_D, 1024, 1.0f, 0);
    // --- 3. V projection, stored transposed (Vt[d][q]) ---
    gemm_nt_bf16<0><<<gProj, blk, 0, stream>>>(xb, 1024, SEQ_D, wvb, 1024, 0,
                                               Vtb, 2048, D_SEQ, 1024, 1.0f, 1);

    // --- 4. scores = (Q @ K^T) / sqrt(1024) ---
    gemm_nt_bf16<0><<<gScore, blk, 0, stream>>>(Qb, 1024, SEQ_D, Kb, 1024, SEQ_D,
                                                Sb, 2048, SEQ_SQ, 1024, 0.03125f, 0);

    // --- 5. softmax over key axis, in place ---
    softmax_rows_bf16<<<4 * 2048, 256, 0, stream>>>(Sb);

    // --- 6. context = P @ V  (== P @ Vt^T) ---
    gemm_nt_bf16<0><<<gProj, blk, 0, stream>>>(Sb, 2048, SEQ_SQ, Vtb, 2048, D_SEQ,
                                               Cb, 1024, SEQ_D, 2048, 1.0f, 0);

    // --- 7. output projection, fp32 result ---
    gemm_nt_bf16<1><<<gProj, blk, 0, stream>>>(Cb, 1024, SEQ_D, wob, 1024, 0,
                                               out, 1024, SEQ_D, 1024, 1.0f, 0);
}
